// LogLinearMamba2Block_62216896250127
// MI455X (gfx1250) — matrix operand order repaired and checked by reading, not compiled, not run on hardware
//
#include <hip/hip_runtime.h>
#include <stdint.h>

typedef __bf16 bf16_t;
typedef __attribute__((ext_vector_type(16))) __bf16 v16bf;
typedef __attribute__((ext_vector_type(8)))  float   v8f;

#define ROWS  2048   // B*T
#define DM    1024
#define TSEQ  1024
#define NPROJ 4368   // 2I + 2GN + H
#define NLAM  176    // H*L
#define CC    2304   // I + 2GN
#define II    2048
#define HH    16
#define NN    128
#define LL    11
#define IMD   2816

// CDNA5 async global->LDS path (guarded so the file compiles on any toolchain)
#if defined(__AMDGCN__) && \
    __has_builtin(__builtin_amdgcn_global_load_async_to_lds_b128) && \
    __has_builtin(__builtin_amdgcn_s_wait_asynccnt)
#define USE_ASYNC_LDS 1
typedef int v4i_async __attribute__((vector_size(16)));
#define AS_GLOBAL_V4(p) \
  ((__attribute__((address_space(1))) v4i_async*)(uintptr_t)(p))
#define AS_LDS_V4(p) \
  ((__attribute__((address_space(3))) v4i_async*)(unsigned)(uintptr_t)(p))
#else
#define USE_ASYNC_LDS 0
#endif

struct __align__(16) U128 { unsigned int x, y, z, w; };

union Frag {
  v16bf v;
  U128 u[2];
  unsigned short s[16];
};

static __device__ __forceinline__ float siluf(float x) {
  return x / (1.0f + expf(-x));
}

// ---------------------------------------------------------------------------
// fp32 [K x N] -> bf16 transposed [N x K]  (weights; makes GEMM B K-contiguous)
// ---------------------------------------------------------------------------
__global__ void cvt_bf16_t_kernel(const float* __restrict__ in,
                                  bf16_t* __restrict__ out, int K, int N) {
  size_t i = (size_t)blockIdx.x * 256 + threadIdx.x;
  if (i < (size_t)K * N) {
    const int k = (int)(i / (unsigned)N);
    const int n = (int)(i % (unsigned)N);
    out[(size_t)n * K + k] = (bf16_t)in[i];
  }
}

// ---------------------------------------------------------------------------
// RMSNorm: fp32 row -> bf16 row (x * rsqrt(mean(x^2)+eps) * w)
// ---------------------------------------------------------------------------
__global__ __launch_bounds__(256) void rmsnorm_bf16_kernel(
    const float* __restrict__ x, const float* __restrict__ w,
    bf16_t* __restrict__ out, int D) {
  __shared__ float red[256];
  const size_t row = blockIdx.x;
  const float* xr = x + row * (size_t)D;
  float ss = 0.f;
  for (int i = threadIdx.x; i < D; i += 256) { float v = xr[i]; ss += v * v; }
  red[threadIdx.x] = ss;
  __syncthreads();
  for (int s = 128; s > 0; s >>= 1) {
    if ((int)threadIdx.x < s) red[threadIdx.x] += red[threadIdx.x + s];
    __syncthreads();
  }
  const float inv = rsqrtf(red[0] / (float)D + 1e-5f);
  for (int i = threadIdx.x; i < D; i += 256)
    out[row * (size_t)D + i] = (bf16_t)(xr[i] * inv * w[i]);
}

// ---------------------------------------------------------------------------
// bf16 GEMM, B pre-transposed: C[M,N] = A[M,K] * Bt[N,K]^T (+ fp32 addend).
// LDS-free; branch-free inner loop (clamped tile columns); SOFTWARE
// DOUBLE-BUFFERED: fragments for K-step i+1 are loaded while step i's WMMAs
// execute, so the load pipe never fully drains. K is a multiple of 128.
// ---------------------------------------------------------------------------
__global__ __launch_bounds__(256) void gemm_bf16_tn_kernel(
    const bf16_t* __restrict__ A, int lda,
    const bf16_t* __restrict__ Bt, int ldb /* = K */,
    float* __restrict__ C, int ldc,
    const float* __restrict__ addend,
    int M, int N, int K) {
  const int wave = threadIdx.x >> 5;
  const int lane = threadIdx.x & 31;
  const int row0 = blockIdx.y * 128 + wave * 16;
  const int col0 = blockIdx.x * 64;

  const v8f vzero = {};
  v8f acc[4];
#pragma unroll
  for (int i = 0; i < 4; ++i) acc[i] = vzero;

  const int kh = (lane >> 4) * 8;   // A: lanes 0-15 K 0..7/16..23, 16-31 +8
  const int kb = (lane >> 4) * 8;   // B: lanes 0-15 K 0..7/16..23, 16-31 +8
  const bf16_t* arow = A + (size_t)(row0 + (lane & 15)) * lda;

  // clamped tile columns (uniform per wave; N is a multiple of 16)
  int nc[4];
  const bf16_t* brow[4];
#pragma unroll
  for (int nt = 0; nt < 4; ++nt) {
    int c = col0 + nt * 16;
    if (c > N - 16) c = N - 16;
    nc[nt] = c;
    brow[nt] = Bt + (size_t)(c + (lane & 15)) * ldb + kb;
  }

  // fragment buffers (double buffered, statically named)
  Frag a0A, a1A, a0B, a1B;
  Frag b0A[4], b1A[4], b0B[4], b1B[4];

  auto loadA = [&](Frag& x0, Frag& x1, int k) {
    x0.u[0] = *(const U128*)(arow + k + kh);
    x0.u[1] = *(const U128*)(arow + k + 16 + kh);
    x1.u[0] = *(const U128*)(arow + k + 32 + kh);
    x1.u[1] = *(const U128*)(arow + k + 48 + kh);
  };
  auto loadB = [&](Frag (&y0)[4], Frag (&y1)[4], int k) {
#pragma unroll
    for (int nt = 0; nt < 4; ++nt) {
      const bf16_t* bp = brow[nt] + k;
      y0[nt].u[0] = *(const U128*)bp;
      y0[nt].u[1] = *(const U128*)(bp + 16);
      y1[nt].u[0] = *(const U128*)(bp + 32);
      y1[nt].u[1] = *(const U128*)(bp + 48);
    }
  };
  auto mmaStep = [&](Frag& x0, Frag& x1, Frag (&y0)[4], Frag (&y1)[4]) {
#pragma unroll
    for (int nt = 0; nt < 4; ++nt) {
      acc[nt] = __builtin_amdgcn_wmma_f32_16x16x32_bf16(
          false, x0.v, false, y0[nt].v, (short)0, acc[nt], false, false);
      acc[nt] = __builtin_amdgcn_wmma_f32_16x16x32_bf16(
          false, x1.v, false, y1[nt].v, (short)0, acc[nt], false, false);
    }
  };

  const int steps = K >> 6;  // K/64; even (K % 128 == 0)
  loadA(a0A, a1A, 0);
  loadB(b0A, b1A, 0);
  for (int it = 0; it < steps; it += 2) {
    const int kA = it << 6;
    // prefetch step it+1 into buffer B while buffer A computes
    loadA(a0B, a1B, kA + 64);
    loadB(b0B, b1B, kA + 64);
    if (kA + 256 < K)  // -> global_prefetch_b8
      __builtin_prefetch((const void*)(arow + kA + 256), 0, 1);
    mmaStep(a0A, a1A, b0A, b1A);
    // prefetch step it+2 into buffer A while buffer B computes
    if (it + 2 < steps) {
      loadA(a0A, a1A, kA + 128);
      loadB(b0A, b1A, kA + 128);
    }
    mmaStep(a0B, a1B, b0B, b1B);
  }

  const int n = lane & 15;
#pragma unroll
  for (int nt = 0; nt < 4; ++nt) {
#pragma unroll
    for (int j = 0; j < 8; ++j) {
      const int r = row0 + j + ((lane >> 4) << 3);
      const size_t off = (size_t)r * ldc + nc[nt] + n;
      float o = acc[nt][j];
      if (addend) o += addend[off];
      C[off] = o;
    }
  }
}

// ---------------------------------------------------------------------------
// Depthwise causal conv (K=4) + bias + SiLU over xBC channels -> bf16
// ---------------------------------------------------------------------------
__global__ void conv_silu_kernel(const float* __restrict__ proj,
                                 const float* __restrict__ cw,
                                 const float* __restrict__ cb,
                                 bf16_t* __restrict__ xbc) {
  const int c = blockIdx.x * 256 + threadIdx.x;
  if (c >= CC) return;
  const int bt = blockIdx.y;
  const int t = bt & (TSEQ - 1);
  float acc = cb[c];
#pragma unroll
  for (int j = 0; j < 4; ++j) {
    const int ts = t - 3 + j;
    if (ts >= 0)
      acc += proj[(size_t)(bt - 3 + j) * NPROJ + II + c] * cw[c * 4 + j];
  }
  xbc[(size_t)bt * CC + c] = (bf16_t)siluf(acc);
}

// ---------------------------------------------------------------------------
// dt = softplus(dt_raw + bias); g = cumsum_t(dt * -exp(A_log))
// One 1024-thread block per (b,h); Hillis-Steele inclusive scan in LDS.
// ---------------------------------------------------------------------------
__global__ __launch_bounds__(1024) void dt_scan_kernel(
    const float* __restrict__ proj, const float* __restrict__ dt_bias,
    const float* __restrict__ A_log, float* __restrict__ dt_out,
    float* __restrict__ g_out) {
  __shared__ float sc[TSEQ];
  const int bh = blockIdx.x;
  const int b = bh >> 4, h = bh & 15;
  const int t = threadIdx.x;
  float dtraw =
      proj[(size_t)(b * TSEQ + t) * NPROJ + (2 * II + 2 * NN) + h] + dt_bias[h];
  float dtv = dtraw > 20.f ? dtraw : log1pf(expf(dtraw));
  const float Ax = -expf(A_log[h]);
  sc[t] = dtv * Ax;
  __syncthreads();
  for (int off = 1; off < TSEQ; off <<= 1) {
    float vprev = (t >= off) ? sc[t - off] : 0.f;
    __syncthreads();
    sc[t] += vprev;
    __syncthreads();
  }
  dt_out[(size_t)bh * TSEQ + t] = dtv;
  g_out[(size_t)bh * TSEQ + t] = sc[t];
}

// ---------------------------------------------------------------------------
// Log-linear attention, flash-style.
// grid = (T/128, B*H), block = 256 (8 waves). Wave w owns 16 query rows.
// Per 32-key block: Bm tile staged to LDS (async-to-LDS when available),
// V tile staged TRANSPOSED ([P][S]) so both the QK B-fragment and the V
// B-fragment are 2x ds_load_b128; qk via WMMA; decay/lambda/dt weighting in
// the fp32 C-layout; re-layout via per-wave LDS; Y += W@V via WMMA.
// ---------------------------------------------------------------------------
__global__ __launch_bounds__(256) void attn_kernel(
    const bf16_t* __restrict__ xbc,   // [B*T, CC] (v | Bm | Cm)
    const float* __restrict__ lam,    // [B*T, H*L]
    const float* __restrict__ gbuf,   // [B*H, T]
    const float* __restrict__ dtbuf,  // [B*H, T]
    const float* __restrict__ Dvec,   // [H]
    float* __restrict__ yout) {       // [B*T, I]
  const int tileT = blockIdx.x;
  const int bh = blockIdx.y;
  const int b = bh >> 4, h = bh & 15;
  const int tBase = tileT * 128;
  const int wave = threadIdx.x >> 5;
  const int lane = threadIdx.x & 31;
  const int t0 = tBase + wave * 16;

  __shared__ __align__(16) bf16_t Bs[32 * 128];    // Bm tile [s][n]
  __shared__ __align__(16) bf16_t Vst[128 * 32];   // V tile transposed [p][s]
  __shared__ __align__(16) bf16_t Wl[8][16 * 32];
  __shared__ float gq[128];
  __shared__ float lamq[128][12];
  __shared__ float gs[32], dts[32];

  if (threadIdx.x < 128)
    gq[threadIdx.x] = gbuf[(size_t)bh * TSEQ + tBase + threadIdx.x];
  for (int i = threadIdx.x; i < 128 * LL; i += 256) {
    const int r = i / LL, l = i % LL;
    lamq[r][l] = lam[(size_t)(b * TSEQ + tBase + r) * NLAM + h * LL + l];
  }

  // preload Cm fragments (A of the QK WMMA)
  Frag cfrag[4];
  {
    const int r = t0 + (lane & 15);
    const int kh = (lane >> 4) * 8;
    const bf16_t* crow = xbc + (size_t)(b * TSEQ + r) * CC + (II + NN);
#pragma unroll
    for (int ks = 0; ks < 4; ++ks) {
      cfrag[ks].u[0] = *(const U128*)(crow + ks * 32 + kh);
      cfrag[ks].u[1] = *(const U128*)(crow + ks * 32 + 16 + kh);
    }
  }

  const v8f vzero = {};
  v8f yacc[8];
#pragma unroll
  for (int i = 0; i < 8; ++i) yacc[i] = vzero;

  const int tMax = tBase + 127;
  for (int s0 = 0; s0 <= tMax; s0 += 32) {
    // ---- cooperative staging of Bm (async-to-LDS) and transposed V ----
    {
      const int base = threadIdx.x * 16;  // 0..4095
      const int r = base >> 7;            // s row 0..31
      const int c = base & 127;           // feature col
      const bf16_t* src = xbc + (size_t)(b * TSEQ + s0 + r) * CC;
#if USE_ASYNC_LDS
      __builtin_amdgcn_global_load_async_to_lds_b128(
          AS_GLOBAL_V4(src + II + c), AS_LDS_V4(&Bs[base]), 0, 0);
      __builtin_amdgcn_global_load_async_to_lds_b128(
          AS_GLOBAL_V4(src + II + c + 8), AS_LDS_V4(&Bs[base + 8]), 0, 0);
#else
      *(U128*)&Bs[base]     = *(const U128*)(src + II + c);
      *(U128*)&Bs[base + 8] = *(const U128*)(src + II + c + 8);
#endif
      // V: coalesced 128-bit global loads, transposed u16 scatter to LDS
      Frag vf;
      vf.u[0] = *(const U128*)(src + h * 128 + c);
      vf.u[1] = *(const U128*)(src + h * 128 + c + 8);
      unsigned short* vt = (unsigned short*)Vst;
#pragma unroll
      for (int e = 0; e < 16; ++e) vt[(c + e) * 32 + r] = vf.s[e];
    }
    if (threadIdx.x < 32) {
      gs[threadIdx.x]  = gbuf[(size_t)bh * TSEQ + s0 + threadIdx.x];
      dts[threadIdx.x] = dtbuf[(size_t)bh * TSEQ + s0 + threadIdx.x];
    }
#if USE_ASYNC_LDS
    __builtin_amdgcn_s_wait_asynccnt(0);
#endif
    __syncthreads();

    const bool active = (s0 <= t0 + 15);
    if (active) {
      // ----- QK: S(16x32) = C(16x128) @ Bm^T(128x32) -----
      v8f S[2];
      S[0] = vzero; S[1] = vzero;
#pragma unroll
      for (int nt = 0; nt < 2; ++nt) {
#pragma unroll
        for (int ks = 0; ks < 4; ++ks) {
          // fragment element (k, n) = Bm[s0 + nt*16 + n, ks*32 + k]
          Frag bfr;
          const bf16_t* bp =
              &Bs[(nt * 16 + (lane & 15)) * 128 + ks * 32 + ((lane >> 4) << 3)];
          bfr.u[0] = *(const U128*)bp;
          bfr.u[1] = *(const U128*)(bp + 16);
          S[nt] = __builtin_amdgcn_wmma_f32_16x16x32_bf16(
              false, cfrag[ks].v, false, bfr.v, (short)0, S[nt], false, false);
        }
      }
      // ----- decay * lambda * dt with causal mask; stage W in LDS -----
#pragma unroll
      for (int nt = 0; nt < 2; ++nt) {
#pragma unroll
        for (int j = 0; j < 8; ++j) {
          const int tl = j + ((lane >> 4) << 3);
          const int t = t0 + tl;
          const int sl = nt * 16 + (lane & 15);
          const int s = s0 + sl;
          float w = 0.f;
          if (s <= t) {
            const int x = t ^ s;
            const int lvl = (x == 0) ? 0 : (32 - __builtin_clz((unsigned)x));
            w = S[nt][j] * expf(gq[t - tBase] - gs[sl]) *
                lamq[t - tBase][lvl] * dts[sl];
          }
          Wl[wave][tl * 32 + sl] = (bf16_t)w;
        }
      }
    }
    __syncthreads();

    if (active) {
      // ----- Y(16x128) += W(16x32) @ V(32x128) -----
      Frag wa;
      {
        const int r = lane & 15;
        const int kh = (lane >> 4) * 8;
        const bf16_t* wp = &Wl[wave][r * 32 + kh];
        wa.u[0] = *(const U128*)wp;
        wa.u[1] = *(const U128*)(wp + 16);
      }
#pragma unroll
      for (int nt = 0; nt < 8; ++nt) {
        // fragment element (k=s, n=p) = Vst[(nt*16+n)*32 + k] : table order
        Frag vb;
        const bf16_t* vp =
            &Vst[(nt * 16 + (lane & 15)) * 32 + ((lane >> 4) << 3)];
        vb.u[0] = *(const U128*)vp;
        vb.u[1] = *(const U128*)(vp + 16);
        yacc[nt] = __builtin_amdgcn_wmma_f32_16x16x32_bf16(
            false, wa.v, false, vb.v, (short)0, yacc[nt], false, false);
      }
    }
    __syncthreads();
  }

  // epilogue: y += D[h] * v ; store fp32
  const float dh = Dvec[h];
#pragma unroll
  for (int nt = 0; nt < 8; ++nt) {
#pragma unroll
    for (int j = 0; j < 8; ++j) {
      const int t = t0 + j + ((lane >> 4) << 3);
      const int p = nt * 16 + (lane & 15);
      const float vv = (float)xbc[(size_t)(b * TSEQ + t) * CC + h * 128 + p];
      yout[(size_t)(b * TSEQ + t) * II + h * 128 + p] = yacc[nt][j] + dh * vv;
    }
  }
}

// ---------------------------------------------------------------------------
// h = y * silu(z); group RMS over I; * gnorm_w -> bf16
// ---------------------------------------------------------------------------
__global__ __launch_bounds__(256) void gate_gnorm_kernel(
    const float* __restrict__ y, const float* __restrict__ proj,
    const float* __restrict__ gw, bf16_t* __restrict__ out) {
  __shared__ float red[256];
  const size_t row = blockIdx.x;
  float hv[8];
  float ss = 0.f;
#pragma unroll
  for (int j = 0; j < 8; ++j) {
    const int i = threadIdx.x + j * 256;
    const float z = proj[row * NPROJ + i];
    const float h = y[row * (size_t)II + i] * siluf(z);
    hv[j] = h;
    ss += h * h;
  }
  red[threadIdx.x] = ss;
  __syncthreads();
  for (int s = 128; s > 0; s >>= 1) {
    if ((int)threadIdx.x < s) red[threadIdx.x] += red[threadIdx.x + s];
    __syncthreads();
  }
  const float inv = rsqrtf(red[0] / (float)II + 1e-5f);
#pragma unroll
  for (int j = 0; j < 8; ++j) {
    const int i = threadIdx.x + j * 256;
    out[row * (size_t)II + i] = (bf16_t)(hv[j] * inv * gw[i]);
  }
}

// ---------------------------------------------------------------------------
// act = silu(gate) * up -> bf16
// ---------------------------------------------------------------------------
__global__ void silu_mul_kernel(const float* __restrict__ g,
                                const float* __restrict__ u,
                                bf16_t* __restrict__ out, size_t n) {
  size_t i = (size_t)blockIdx.x * 256 + threadIdx.x;
  if (i < n) out[i] = (bf16_t)(siluf(g[i]) * u[i]);
}

// ---------------------------------------------------------------------------
extern "C" void kernel_launch(void* const* d_in, const int* in_sizes, int n_in,
                              void* d_out, int out_size, void* d_ws,
                              size_t ws_size, hipStream_t stream) {
  const float* hidden       = (const float*)d_in[0];
  const float* mixer_norm_w = (const float*)d_in[1];
  const float* mlp_norm_w   = (const float*)d_in[2];
  const float* in_proj_w    = (const float*)d_in[3];
  const float* conv_w       = (const float*)d_in[4];
  const float* conv_b       = (const float*)d_in[5];
  const float* dt_bias      = (const float*)d_in[6];
  const float* A_log        = (const float*)d_in[7];
  const float* Dv           = (const float*)d_in[8];
  const float* lam_w        = (const float*)d_in[9];
  const float* gnorm_w      = (const float*)d_in[10];
  const float* out_proj_w   = (const float*)d_in[11];
  const float* gate_w       = (const float*)d_in[12];
  const float* up_w         = (const float*)d_in[13];
  const float* down_w       = (const float*)d_in[14];
  float* outp = (float*)d_out;

  char* cur = (char*)d_ws;
  auto carve = [&cur](size_t bytes) -> void* {
    char* r = cur;
    cur += (bytes + 255) & ~(size_t)255;
    return (void*)r;
  };

  bf16_t* x_bf      = (bf16_t*)carve((size_t)ROWS * DM * 2);
  bf16_t* w_in_t    = (bf16_t*)carve((size_t)DM * NPROJ * 2);   // [N][K]
  bf16_t* w_lam_t   = (bf16_t*)carve((size_t)DM * NLAM * 2);
  bf16_t* w_out_t   = (bf16_t*)carve((size_t)II * DM * 2);
  bf16_t* w_gate_t  = (bf16_t*)carve((size_t)DM * IMD * 2);
  bf16_t* w_up_t    = (bf16_t*)carve((size_t)DM * IMD * 2);
  bf16_t* w_down_t  = (bf16_t*)carve((size_t)IMD * DM * 2);
  float*  projb     = (float*)carve((size_t)ROWS * NPROJ * 4);
  float*  lamb      = (float*)carve((size_t)ROWS * NLAM * 4);
  bf16_t* xbc       = (bf16_t*)carve((size_t)ROWS * CC * 2);
  float*  dtb       = (float*)carve((size_t)32 * TSEQ * 4);
  float*  gb        = (float*)carve((size_t)32 * TSEQ * 4);
  float*  yb        = (float*)carve((size_t)ROWS * II * 4);
  bf16_t* h_bf      = (bf16_t*)carve((size_t)ROWS * II * 2);
  float*  res2      = (float*)carve((size_t)ROWS * DM * 4);
  bf16_t* hn_bf     = (bf16_t*)carve((size_t)ROWS * DM * 2);
  float*  gateb     = (float*)carve((size_t)ROWS * IMD * 4);
  float*  upb       = (float*)carve((size_t)ROWS * IMD * 4);
  bf16_t* act_bf    = (bf16_t*)carve((size_t)ROWS * IMD * 2);

  auto cvt_t = [&](const float* src, bf16_t* dst, int K, int N) {
    const size_t n = (size_t)K * N;
    cvt_bf16_t_kernel<<<dim3((unsigned)((n + 255) / 256)), dim3(256), 0,
                        stream>>>(src, dst, K, N);
  };
  cvt_t(in_proj_w, w_in_t, DM, NPROJ);
  cvt_t(lam_w, w_lam_t, DM, NLAM);
  cvt_t(out_proj_w, w_out_t, II, DM);
  cvt_t(gate_w, w_gate_t, DM, IMD);
  cvt_t(up_w, w_up_t, DM, IMD);
  cvt_t(down_w, w_down_t, IMD, DM);

  // mixer RMS norm -> bf16 activations
  rmsnorm_bf16_kernel<<<dim3(ROWS), dim3(256), 0, stream>>>(
      hidden, mixer_norm_w, x_bf, DM);

  // in_proj and lambda projections
  gemm_bf16_tn_kernel<<<dim3((NPROJ + 63) / 64, ROWS / 128), dim3(256), 0,
                        stream>>>(x_bf, DM, w_in_t, DM, projb, NPROJ, nullptr,
                                  ROWS, NPROJ, DM);
  gemm_bf16_tn_kernel<<<dim3((NLAM + 63) / 64, ROWS / 128), dim3(256), 0,
                        stream>>>(x_bf, DM, w_lam_t, DM, lamb, NLAM, nullptr,
                                  ROWS, NLAM, DM);

  // conv + silu, dt/g scan
  conv_silu_kernel<<<dim3((CC + 255) / 256, ROWS), dim3(256), 0, stream>>>(
      projb, conv_w, conv_b, xbc);
  dt_scan_kernel<<<dim3(32), dim3(1024), 0, stream>>>(projb, dt_bias, A_log,
                                                      dtb, gb);

  // log-linear attention
  attn_kernel<<<dim3(TSEQ / 128, 32), dim3(256), 0, stream>>>(xbc, lamb, gb,
                                                              dtb, Dv, yb);

  // gating + group norm -> bf16
  gate_gnorm_kernel<<<dim3(ROWS), dim3(256), 0, stream>>>(yb, projb, gnorm_w,
                                                          h_bf);

  // out_proj with fused residual add (res2 = residual + mixer_out)
  gemm_bf16_tn_kernel<<<dim3((DM + 63) / 64, ROWS / 128), dim3(256), 0,
                        stream>>>(h_bf, II, w_out_t, II, res2, DM, hidden,
                                  ROWS, DM, II);

  // MLP norm -> bf16
  rmsnorm_bf16_kernel<<<dim3(ROWS), dim3(256), 0, stream>>>(res2, mlp_norm_w,
                                                            hn_bf, DM);

  // gate / up projections
  gemm_bf16_tn_kernel<<<dim3((IMD + 63) / 64, ROWS / 128), dim3(256), 0,
                        stream>>>(hn_bf, DM, w_gate_t, DM, gateb, IMD, nullptr,
                                  ROWS, IMD, DM);
  gemm_bf16_tn_kernel<<<dim3((IMD + 63) / 64, ROWS / 128), dim3(256), 0,
                        stream>>>(hn_bf, DM, w_up_t, DM, upb, IMD, nullptr,
                                  ROWS, IMD, DM);

  // silu(gate)*up -> bf16
  silu_mul_kernel<<<dim3((unsigned)(((size_t)ROWS * IMD + 255) / 256)),
                    dim3(256), 0, stream>>>(gateb, upb, act_bf,
                                            (size_t)ROWS * IMD);

  // down projection with fused residual (out = res2 + mlp)
  gemm_bf16_tn_kernel<<<dim3((DM + 63) / 64, ROWS / 128), dim3(256), 0,
                        stream>>>(act_bf, IMD, w_down_t, IMD, outp, DM, res2,
                                  ROWS, DM, IMD);

  (void)in_sizes; (void)n_in; (void)out_size; (void)ws_size;
}
